// GNNBranch_68839735820749
// MI455X (gfx1250) — matrix operand order repaired and checked by reading, not compiled, not run on hardware
//
#include <hip/hip_runtime.h>

typedef __attribute__((ext_vector_type(16))) _Float16 v16h;
typedef __attribute__((ext_vector_type(8)))  float    v8f;

#define NEG_SLOPE 0.01f
#define LN_EPS    1e-5f
#define WAVES     8   // 256-thread block = 8 wave32

// ---------------------------------------------------------------------------
// WMMA helpers — CDNA5 16x16x32 f16 -> f32, layouts per ISA 7.12.2 (wave32)
// ---------------------------------------------------------------------------
__device__ __forceinline__ v8f wmma_f16(v16h a, v16h b, v8f c) {
  return __builtin_amdgcn_wmma_f32_16x16x32_f16(false, a, false, b, (short)0, c,
                                                false, false);
}

// K index of element e (0..15) of a v16h A-fragment (16x32, M x K), hi = lane>>4
__device__ __forceinline__ int a_k(int e, int hi) {
  return ((e & 8) << 1) + (((e >> 1) & 3) << 1) + (hi << 3) + (e & 1);
}

// A fragment from row-major f32 global memory (row m of this tile, ld floats/row)
__device__ __forceinline__ v16h load_a_g32(const float* __restrict__ base, int ld,
                                           int m, int hi, int kBase) {
  v16h a;
#pragma unroll
  for (int e = 0; e < 16; e += 2) {          // K pairs are contiguous -> float2
    int k = kBase + a_k(e, hi);
    float2 p = *(const float2*)(base + (size_t)m * ld + k);
    a[e]     = (_Float16)p.x;
    a[e + 1] = (_Float16)p.y;
  }
  return a;
}

// A fragment from an f16 LDS tile, zero-padded beyond kLimit (layer-2 K=16 pads)
__device__ __forceinline__ v16h load_a_lds(const _Float16* t, int ld, int m, int hi,
                                           int kBase, int kLimit) {
  v16h a;
#pragma unroll
  for (int e = 0; e < 16; ++e) {
    int k = kBase + a_k(e, hi);
    a[e] = (k < kLimit) ? t[m * ld + k] : (_Float16)0.0f;
  }
  return a;
}

// B fragment (32x16, K x N) from row-major f32 weight [kTot x ncols]; zero pad K>=kTot
__device__ __forceinline__ v16h load_b_g32(const float* __restrict__ w, int ncols,
                                           int nBase, int kBase, int kTot, int lane) {
  int n = lane & 15, hi = lane >> 4;
  v16h b;
#pragma unroll
  for (int e = 0; e < 16; ++e) {
    int k = kBase + ((e & 8) << 1) + (hi << 3) + (e & 7);
    b[e] = (k < kTot) ? (_Float16)w[(size_t)k * ncols + nBase + n] : (_Float16)0.0f;
  }
  return b;
}

__device__ __forceinline__ v8f bias_lrelu(v8f c, float bias) {
#pragma unroll
  for (int i = 0; i < 8; ++i) {
    float v = c[i] + bias;
    c[i] = v > 0.0f ? v : NEG_SLOPE * v;
  }
  return c;
}

// D tile (16x16 f32, C/D layout) -> f16 LDS tile [16 x ld] at column colBase
__device__ __forceinline__ void store_d_lds(_Float16* t, int ld, int colBase,
                                            int lane, v8f d) {
  int n = lane & 15, hi = lane >> 4;
#pragma unroll
  for (int r = 0; r < 8; ++r)
    t[(r + (hi << 3)) * ld + colBase + n] = (_Float16)d[r];
}

// ---------------------------------------------------------------------------
// Kernel 0: zero the scatter accumulator
// ---------------------------------------------------------------------------
__global__ void zero_f32(float* __restrict__ p, size_t n) {
  size_t i = (size_t)blockIdx.x * blockDim.x + threadIdx.x;
  size_t s = (size_t)gridDim.x * blockDim.x;
  for (; i < n; i += s) p[i] = 0.0f;
}

// ---------------------------------------------------------------------------
// Kernel 1: node encoder  x[N,128] -> lrelu(lrelu(x@w1+b1)@w2+b2) -> f16 [N,32]
// ---------------------------------------------------------------------------
__global__ __launch_bounds__(256)
void node_enc_kernel(const float* __restrict__ x,
                     const float* __restrict__ w1, const float* __restrict__ b1,
                     const float* __restrict__ w2, const float* __restrict__ b2,
                     _Float16* __restrict__ node_enc, int N, int nTiles) {
  __shared__ _Float16 sTmp[WAVES][16 * 16];
  const int lane = threadIdx.x & 31, wave = threadIdx.x >> 5;
  const int n = lane & 15, hi = lane >> 4, m = n;
  _Float16* tmp = sTmp[wave];

  v16h B1[4], B2[2];
#pragma unroll
  for (int kb = 0; kb < 4; ++kb) B1[kb] = load_b_g32(w1, 16, 0, kb * 32, 128, lane);
#pragma unroll
  for (int nt = 0; nt < 2; ++nt)  B2[nt] = load_b_g32(w2, 32, nt * 16, 0, 16, lane);
  const float bi1 = b1[n];
  const float bi2[2] = {b2[n], b2[16 + n]};

  const int waveStride = gridDim.x * WAVES;
  for (int tile = blockIdx.x * WAVES + wave; tile < nTiles; tile += waveStride) {
    const int row0 = tile * 16;
    const float* xr = x + (size_t)row0 * 128;

    v8f acc = {};
#pragma unroll
    for (int kb = 0; kb < 4; ++kb)
      acc = wmma_f16(load_a_g32(xr, 128, m, hi, kb * 32), B1[kb], acc);
    acc = bias_lrelu(acc, bi1);
    store_d_lds(tmp, 16, 0, lane, acc);

    v16h Ah = load_a_lds(tmp, 16, m, hi, 0, 16);
#pragma unroll
    for (int nt = 0; nt < 2; ++nt) {
      v8f z = {};
      v8f o = bias_lrelu(wmma_f16(Ah, B2[nt], z), bi2[nt]);
#pragma unroll
      for (int r = 0; r < 8; ++r) {
        int grow = row0 + r + (hi << 3);
        if (grow < N) node_enc[(size_t)grow * 32 + nt * 16 + n] = (_Float16)o[r];
      }
    }
  }
}

// ---------------------------------------------------------------------------
// Kernel 2: fused edge pipeline: edge encoder + gather node_enc[src] +
//           message MLP + atomic scatter-add into gnn_out[N,64]
// ---------------------------------------------------------------------------
__global__ __launch_bounds__(256)
void edge_kernel(const float* __restrict__ edge_attr,
                 const int* __restrict__ ei,                 // [2,E] flat
                 const _Float16* __restrict__ node_enc,      // [N,32] f16
                 const float* __restrict__ ee_w1, const float* __restrict__ ee_b1,
                 const float* __restrict__ ee_w2, const float* __restrict__ ee_b2,
                 const float* __restrict__ gn_w1, const float* __restrict__ gn_b1,
                 const float* __restrict__ gn_w2, const float* __restrict__ gn_b2,
                 float* __restrict__ gnn_out, int E, int eTiles) {
  __shared__ _Float16 sFeats[WAVES][16 * 64];  // [edge_enc | node_enc[src]]
  __shared__ _Float16 sTmp[WAVES][16 * 16];
  __shared__ int      sDst[WAVES][16];
  const int lane = threadIdx.x & 31, wave = threadIdx.x >> 5;
  const int n = lane & 15, hi = lane >> 4, m = n;
  _Float16* feats = sFeats[wave];
  _Float16* tmp   = sTmp[wave];
  const int* srcI = ei;
  const int* dstI = ei + E;

  v16h Bee1[2], Bee2[2], Bgn1[2], Bgn2[4];
#pragma unroll
  for (int kb = 0; kb < 2; ++kb) Bee1[kb] = load_b_g32(ee_w1, 16, 0, kb * 32, 64, lane);
#pragma unroll
  for (int nt = 0; nt < 2; ++nt) Bee2[nt] = load_b_g32(ee_w2, 32, nt * 16, 0, 16, lane);
#pragma unroll
  for (int kb = 0; kb < 2; ++kb) Bgn1[kb] = load_b_g32(gn_w1, 16, 0, kb * 32, 64, lane);
#pragma unroll
  for (int nt = 0; nt < 4; ++nt) Bgn2[nt] = load_b_g32(gn_w2, 64, nt * 16, 0, 16, lane);
  const float be1 = ee_b1[n];
  const float be2[2] = {ee_b2[n], ee_b2[16 + n]};
  const float bg1 = gn_b1[n];
  const float bg2[4] = {gn_b2[n], gn_b2[16 + n], gn_b2[32 + n], gn_b2[48 + n]};

  const int waveStride = gridDim.x * WAVES;
  for (int tile = blockIdx.x * WAVES + wave; tile < eTiles; tile += waveStride) {
    const int e0 = tile * 16;

    // stage dst indices (lanes 0..15), gather node_enc[src] rows into feats[:,32:64]
    if (lane < 16) sDst[wave][lane] = dstI[min(e0 + lane, E - 1)];
    {
      int srow = srcI[min(e0 + m, E - 1)];
      const int4* gp = (const int4*)(node_enc + (size_t)srow * 32 + (hi << 4));
      int4* lp = (int4*)(feats + m * 64 + 32 + (hi << 4));
      lp[0] = gp[0];                       // 8 halves
      lp[1] = gp[1];                       // 8 halves -> 16 cols per lane
    }

    // prefetch next tile's edge_attr slab (4 KB, one 128B chunk per lane)
    int nxt = tile + waveStride;
    if (nxt < eTiles)
      __builtin_prefetch(edge_attr + (size_t)nxt * 16 * 64 + lane * 32, 0, 1);

    // ---- edge encoder layer 1: [16x64] @ [64x16]
    const float* ea = edge_attr + (size_t)e0 * 64;
    v8f a1 = {};
#pragma unroll
    for (int kb = 0; kb < 2; ++kb)
      a1 = wmma_f16(load_a_g32(ea, 64, m, hi, kb * 32), Bee1[kb], a1);
    a1 = bias_lrelu(a1, be1);
    store_d_lds(tmp, 16, 0, lane, a1);

    // ---- edge encoder layer 2: [16x16] @ [16x32] -> feats[:,0:32]
    v16h Ah = load_a_lds(tmp, 16, m, hi, 0, 16);
#pragma unroll
    for (int nt = 0; nt < 2; ++nt) {
      v8f z = {};
      v8f t = bias_lrelu(wmma_f16(Ah, Bee2[nt], z), be2[nt]);
      store_d_lds(feats, 64, nt * 16, lane, t);
    }

    // ---- message MLP layer 1: [16x64] @ [64x16]
    v8f g1 = {};
#pragma unroll
    for (int kb = 0; kb < 2; ++kb)
      g1 = wmma_f16(load_a_lds(feats, 64, m, hi, kb * 32, 64), Bgn1[kb], g1);
    g1 = bias_lrelu(g1, bg1);
    store_d_lds(tmp, 16, 0, lane, g1);

    // ---- message MLP layer 2 + scatter-add (atomics resolve in L2)
    v16h Ag = load_a_lds(tmp, 16, m, hi, 0, 16);
#pragma unroll
    for (int nt = 0; nt < 4; ++nt) {
      v8f z = {};
      v8f msg = bias_lrelu(wmma_f16(Ag, Bgn2[nt], z), bg2[nt]);
#pragma unroll
      for (int r = 0; r < 8; ++r) {
        int em = r + (hi << 3);
        if (e0 + em < E) {
          int d = sDst[wave][em];
          atomicAdd(&gnn_out[(size_t)d * 64 + nt * 16 + n], msg[r]);
        }
      }
    }
  }
}

// ---------------------------------------------------------------------------
// Kernel 3: LayerNorm(concat[gnn_out, x]) + 192->32->32->64 MLP
// ---------------------------------------------------------------------------
__global__ __launch_bounds__(256)
void post_kernel(const float* __restrict__ x, const float* __restrict__ gnn_out,
                 const float* __restrict__ ln_g, const float* __restrict__ ln_b,
                 const float* __restrict__ w1, const float* __restrict__ b1,
                 const float* __restrict__ w2, const float* __restrict__ b2,
                 const float* __restrict__ w3, const float* __restrict__ b3,
                 float* __restrict__ out, int N, int nTiles) {
  __shared__ _Float16 sH[WAVES][16 * 192];
  __shared__ _Float16 sTmp[WAVES][16 * 32];
  const int lane = threadIdx.x & 31, wave = threadIdx.x >> 5;
  const int n = lane & 15, hi = lane >> 4, m = n;
  _Float16* h   = sH[wave];
  _Float16* tmp = sTmp[wave];

  v16h B1[2][6], B2[2], B3[4];
#pragma unroll
  for (int nt = 0; nt < 2; ++nt)
#pragma unroll
    for (int kb = 0; kb < 6; ++kb)
      B1[nt][kb] = load_b_g32(w1, 32, nt * 16, kb * 32, 192, lane);
#pragma unroll
  for (int nt = 0; nt < 2; ++nt) B2[nt] = load_b_g32(w2, 32, nt * 16, 0, 32, lane);
#pragma unroll
  for (int nt = 0; nt < 4; ++nt) B3[nt] = load_b_g32(w3, 64, nt * 16, 0, 32, lane);
  const float bi1[2] = {b1[n], b1[16 + n]};
  const float bi2[2] = {b2[n], b2[16 + n]};
  const float bi3[4] = {b3[n], b3[16 + n], b3[32 + n], b3[48 + n]};

  const int waveStride = gridDim.x * WAVES;
  for (int tile = blockIdx.x * WAVES + wave; tile < nTiles; tile += waveStride) {
    const int row0 = tile * 16;
    const int gr = min(row0 + m, N - 1);

    // LayerNorm over 192 feats: lane covers 96, partner (lane^16) the other 96
    float s = 0.0f, sq = 0.0f;
    for (int f = hi * 96; f < hi * 96 + 96; ++f) {
      float v = (f < 64) ? gnn_out[(size_t)gr * 64 + f]
                         : x[(size_t)gr * 128 + (f - 64)];
      s += v; sq += v * v;
    }
    s  += __shfl_xor(s, 16, 32);
    sq += __shfl_xor(sq, 16, 32);
    const float mean = s * (1.0f / 192.0f);
    const float var  = sq * (1.0f / 192.0f) - mean * mean;
    const float rs   = rsqrtf(var + LN_EPS);
    for (int f = hi * 96; f < hi * 96 + 96; ++f) {
      float v = (f < 64) ? gnn_out[(size_t)gr * 64 + f]
                         : x[(size_t)gr * 128 + (f - 64)];
      h[m * 192 + f] = (_Float16)((v - mean) * rs * ln_g[f] + ln_b[f]);
    }

    // layer 1: [16x192] @ [192x32]
    v16h Af[6];
#pragma unroll
    for (int kb = 0; kb < 6; ++kb) Af[kb] = load_a_lds(h, 192, m, hi, kb * 32, 192);
#pragma unroll
    for (int nt = 0; nt < 2; ++nt) {
      v8f a = {};
#pragma unroll
      for (int kb = 0; kb < 6; ++kb) a = wmma_f16(Af[kb], B1[nt][kb], a);
      store_d_lds(tmp, 32, nt * 16, lane, bias_lrelu(a, bi1[nt]));
    }

    // layer 2: [16x32] @ [32x32]
    v16h A2 = load_a_lds(tmp, 32, m, hi, 0, 32);
#pragma unroll
    for (int nt = 0; nt < 2; ++nt) {
      v8f z = {};
      store_d_lds(tmp, 32, nt * 16, lane, bias_lrelu(wmma_f16(A2, B2[nt], z), bi2[nt]));
    }

    // layer 3 (no activation): [16x32] @ [32x64] -> out
    v16h A3 = load_a_lds(tmp, 32, m, hi, 0, 32);
#pragma unroll
    for (int nt = 0; nt < 4; ++nt) {
      v8f z = {};
      v8f o = wmma_f16(A3, B3[nt], z);
#pragma unroll
      for (int r = 0; r < 8; ++r) {
        int grow = row0 + r + (hi << 3);
        if (grow < N) out[(size_t)grow * 64 + nt * 16 + n] = o[r] + bi3[nt];
      }
    }
  }
}

// ---------------------------------------------------------------------------
extern "C" void kernel_launch(void* const* d_in, const int* in_sizes, int n_in,
                              void* d_out, int out_size, void* d_ws, size_t ws_size,
                              hipStream_t stream) {
  const float* x         = (const float*)d_in[0];
  const int*   ei        = (const int*)d_in[1];
  const float* edge_attr = (const float*)d_in[2];
  const float* ne_w1 = (const float*)d_in[3];
  const float* ne_b1 = (const float*)d_in[4];
  const float* ne_w2 = (const float*)d_in[5];
  const float* ne_b2 = (const float*)d_in[6];
  const float* ee_w1 = (const float*)d_in[7];
  const float* ee_b1 = (const float*)d_in[8];
  const float* ee_w2 = (const float*)d_in[9];
  const float* ee_b2 = (const float*)d_in[10];
  const float* gn_w1 = (const float*)d_in[11];
  const float* gn_b1 = (const float*)d_in[12];
  const float* gn_w2 = (const float*)d_in[13];
  const float* gn_b2 = (const float*)d_in[14];
  const float* ln_g  = (const float*)d_in[15];
  const float* ln_b  = (const float*)d_in[16];
  const float* pp_w1 = (const float*)d_in[17];
  const float* pp_b1 = (const float*)d_in[18];
  const float* pp_w2 = (const float*)d_in[19];
  const float* pp_b2 = (const float*)d_in[20];
  const float* pp_w3 = (const float*)d_in[21];
  const float* pp_b3 = (const float*)d_in[22];

  const int N = in_sizes[0] / 128;
  const int E = in_sizes[1] / 2;
  const int nTiles = (N + 15) / 16;
  const int eTiles = (E + 15) / 16;

  // Workspace: node_enc f16 [N,32] then gnn_out f32 [N,64] (re-zeroed per launch)
  _Float16* node_enc = (_Float16*)d_ws;
  size_t neBytes = ((size_t)N * 32 * sizeof(_Float16) + 255) & ~(size_t)255;
  float* gnn = (float*)((char*)d_ws + neBytes);

  zero_f32<<<1024, 256, 0, stream>>>(gnn, (size_t)N * 64);

  int nBlocks = (nTiles + WAVES - 1) / WAVES;
  if (nBlocks > 2048) nBlocks = 2048;
  node_enc_kernel<<<nBlocks, 256, 0, stream>>>(x, ne_w1, ne_b1, ne_w2, ne_b2,
                                               node_enc, N, nTiles);

  int eBlocks = (eTiles + WAVES - 1) / WAVES;
  if (eBlocks > 4096) eBlocks = 4096;
  edge_kernel<<<eBlocks, 256, 0, stream>>>(edge_attr, ei, node_enc,
                                           ee_w1, ee_b1, ee_w2, ee_b2,
                                           gn_w1, gn_b1, gn_w2, gn_b2,
                                           gnn, E, eTiles);

  post_kernel<<<nBlocks, 256, 0, stream>>>(x, gnn, ln_g, ln_b,
                                           pp_w1, pp_b1, pp_w2, pp_b2, pp_w3, pp_b3,
                                           (float*)d_out, N, nTiles);
}
